// ResonantAttentionHeadV2_75892072121061
// MI455X (gfx1250) — matrix operand order repaired and checked by reading, not compiled, not run on hardware
//
#include <hip/hip_runtime.h>
#include <hip/hip_bf16.h>
#include <math.h>

typedef __attribute__((ext_vector_type(16))) _Float16 v16h;
typedef __attribute__((ext_vector_type(8)))  float    v8f;
typedef __attribute__((ext_vector_type(2)))  float    v2f;
typedef __attribute__((ext_vector_type(4)))  unsigned u32x4;
typedef __attribute__((ext_vector_type(8)))  int      i32x8;
typedef __attribute__((ext_vector_type(4)))  int      i32x4;

#define Bsz 16
#define Hn  4096
#define Dn  4096
#define DHn 128

#define F_TWO_PI 6.28318530717958647692f
#define F_PHI    1.61803398874989484820f

#if __has_builtin(__builtin_amdgcn_wmma_f32_16x16x4_f32)
#define HAVE_WMMA_F32X4 1
#else
#warning "CDNA5 probe: __builtin_amdgcn_wmma_f32_16x16x4_f32 NOT available -> FMA fallback in output_kernel"
#endif

#if __has_builtin(__builtin_amdgcn_tensor_load_to_lds) && __has_builtin(__builtin_amdgcn_s_wait_tensorcnt)
#define HAVE_TDM 1
#else
#warning "CDNA5 probe: tensor_load_to_lds builtin NOT available -> manual global->LDS staging"
#endif

// Faithful re-implementation of the reference 4096-entry LUT with linear interp.
__device__ __forceinline__ void lut_sincos(float theta, float& s_out, float& c_out) {
    const float inv_step = 4096.0f / F_TWO_PI;
    const float step     = F_TWO_PI / 4096.0f;
    float pos = fmodf(theta, F_TWO_PI);
    if (pos < 0.0f) pos += F_TWO_PI;          // jnp.mod: non-negative for positive divisor
    float idx_f = pos * inv_step;
    float i0f   = floorf(idx_f);
    float frac  = idx_f - i0f;
    int i0 = ((int)i0f) & 4095;
    int i1 = (i0 + 1) & 4095;
    float g0 = (float)i0 * step;
    float g1 = (float)i1 * step;
    float s0 = __sinf(g0), s1 = __sinf(g1);
    float c0 = __cosf(g0), c1 = __cosf(g1);
    s_out = s0 * (1.0f - frac) + s1 * frac;
    c_out = c0 * (1.0f - frac) + c1 * frac;
}

// ---------------------------------------------------------------------------
// Phase 1: q2[b][0:128]=cos(theta_q), q2[b][128:256]=sin(theta_q)   (tiny)
// ---------------------------------------------------------------------------
__global__ void compute_q_kernel(const float* __restrict__ x,
                                 const float* __restrict__ t,
                                 const float* __restrict__ wq,
                                 const float* __restrict__ bq,
                                 float* __restrict__ q2) {
    int i = blockIdx.x * blockDim.x + threadIdx.x;
    if (i >= Bsz * DHn) return;
    int b = i / DHn, j = i % DHn;
    float theta = x[b * Dn + j] / (1.0f + fabsf(wq[j])) + bq[j] + t[b] * F_PHI;
    float s, c;
    lut_sincos(theta, s, c);
    q2[b * 256 + j]       = c;   // cq
    q2[b * 256 + 128 + j] = s;   // sq
}

// ---------------------------------------------------------------------------
// Phase 2: scores[b][h] = dot(q2[b], k2[b][h]) / 16 via v_wmma_f32_16x16x32_f16.
// One wave per 16-h tile. The 16x128 f32 tile (row stride 4096 f32) is DMA'd
// into LDS by the Tensor Data Mover, then each 32-wide column chunk feeds TWO
// WMMAs (cos slab vs q[0:128], sin slab vs q[128:256]).
// ---------------------------------------------------------------------------
__global__ void __launch_bounds__(256) scores_kernel(const float* __restrict__ cs,
                                                     const float* __restrict__ wk,
                                                     const float* __restrict__ bk,
                                                     const float* __restrict__ q2,
                                                     float* __restrict__ scores) {
    __shared__ float    kraw[8][16 * 128];   // raw tile per wave (8 KB each)
    __shared__ _Float16 kstC[8][512];        // cos B-staging: [K=32][N=16]
    __shared__ _Float16 kstS[8][512];        // sin B-staging
    __shared__ _Float16 q2h[256];

    const int tid  = threadIdx.x;
    const int lane = tid & 31;
    const int w    = tid >> 5;
    const int b    = blockIdx.y;
    const int hi   = (lane >= 16);
    const int nsel = lane & 15;

    q2h[tid] = (_Float16)q2[b * 256 + tid];
    __syncthreads();

    const int h0 = (blockIdx.x * 8 + w) * 16;
    const float* tile_base = cs + ((size_t)b * Hn + h0) * Dn;   // cols 0..127 used

#ifdef HAVE_TDM
    {
        // Build the Tensor DMA Descriptor (D#), ISA cdna5 ch.8.
        // 2D tile: tile_dim0=128 elems (contig), tile_dim1=16 rows,
        // tensor_dim0=4096, tensor_dim0_stride=4096, data_size=4B.
        unsigned lds_off = (unsigned)(size_t)(&kraw[w][0]);      // low32 of flat addr = LDS offset
        unsigned long long ga = (unsigned long long)(size_t)tile_base;

        u32x4 g0;
        g0[0] = (unsigned)__builtin_amdgcn_readfirstlane(1);                 // count=1, user mode
        g0[1] = (unsigned)__builtin_amdgcn_readfirstlane((int)lds_off);      // lds_addr
        g0[2] = (unsigned)__builtin_amdgcn_readfirstlane((int)(ga & 0xFFFFFFFFu));
        g0[3] = (unsigned)__builtin_amdgcn_readfirstlane(
                    (int)(((unsigned)((ga >> 32) & 0x01FFFFFFull)) | 0x80000000u)); // type=2

        i32x8 g1;
        g1[0] = __builtin_amdgcn_readfirstlane((int)0x00020000u);  // data_size=2 (4B), wg_mask=0
        g1[1] = __builtin_amdgcn_readfirstlane((int)(4096u << 16)); // tensor_dim0 lo16 @bits63:48
        g1[2] = __builtin_amdgcn_readfirstlane((int)(16u << 16));   // tdim0 hi16=0 | tensor_dim1 lo16=16
        g1[3] = __builtin_amdgcn_readfirstlane((int)(128u << 16));  // tdim1 hi16=0 | tile_dim0=128
        g1[4] = __builtin_amdgcn_readfirstlane(16);                 // tile_dim1=16, tile_dim2=0
        g1[5] = __builtin_amdgcn_readfirstlane(4096);               // tensor_dim0_stride lo32
        g1[6] = 0;
        g1[7] = 0;

        i32x4 gz = {0, 0, 0, 0};
#if __clang_major__ >= 23
        i32x8 gz8 = {0, 0, 0, 0, 0, 0, 0, 0};
        __builtin_amdgcn_tensor_load_to_lds(g0, g1, gz, gz, gz8, 0);
#else
        __builtin_amdgcn_tensor_load_to_lds(g0, g1, gz, gz, 0);
#endif
        __builtin_amdgcn_s_wait_tensorcnt(0);
        asm volatile("" ::: "memory");
    }
#else
    // Manual coalesced staging: each lane loads 64 floats (128B per row step).
    for (int i = lane; i < 16 * 128; i += 32) {
        int n = i >> 7, col = i & 127;
        kraw[w][i] = tile_base[(size_t)n * Dn + col];
    }
    asm volatile("s_wait_dscnt 0" ::: "memory");
#endif

    v8f acc = {};

    for (int cc = 0; cc < 4; ++cc) {
        const int jj = cc * 32 + lane;                 // feature column 0..127
        const float invw = 1.0f / (1.0f + fabsf(wk[jj]));
        const float bkj  = bk[jj];

        // lane == K index within chunk; produce 16 N values, both cos and sin
        #pragma unroll
        for (int n = 0; n < 16; ++n) {
            float theta = kraw[w][n * 128 + jj] * invw + bkj;
            float s, c;
            lut_sincos(theta, s, c);
            kstC[w][lane * 16 + n] = (_Float16)c;
            kstS[w][lane * 16 + n] = (_Float16)s;
        }
        // same-wave DS ops are in-order; keep the compiler honest too
        asm volatile("s_wait_dscnt 0" ::: "memory");

        v16h aC, bC, aS, bS;
        #pragma unroll
        for (int e = 0; e < 16; ++e) {
            // A 16x32 f16, contiguous k order used here: lanes0-15 hold K=0..15, lanes16-31 K=16..31
            int ka = e + (hi ? 16 : 0);
            aC[e] = q2h[cc * 32 + ka];          // cos slab of q  (broadcast over M)
            aS[e] = q2h[128 + cc * 32 + ka];    // sin slab of q
            // B 32x16 f16, same k order as A: lanes0-15 hold K=0..15, lanes16-31 K=16..31
            int kb = e + (hi ? 16 : 0);
            bC[e] = kstC[w][kb * 16 + nsel];
            bS[e] = kstS[w][kb * 16 + nsel];
        }
        acc = __builtin_amdgcn_wmma_f32_16x16x32_f16(false, aC, false, bC,
                                                     (short)0, acc, false, false);
        acc = __builtin_amdgcn_wmma_f32_16x16x32_f16(false, aS, false, bS,
                                                     (short)0, acc, false, false);
    }
    // D row M=0 lives in acc[0] of lanes 0..15 (all rows identical by construction)
    if (lane < 16)
        scores[b * Hn + h0 + lane] = acc[0] * (1.0f / 16.0f);   // /= sqrt(2*DH)
}

// ---------------------------------------------------------------------------
// Phase 3: softmax over h=4096 per batch
// ---------------------------------------------------------------------------
__global__ void __launch_bounds__(1024) softmax_kernel(const float* __restrict__ scores,
                                                       float* __restrict__ weights) {
    __shared__ float red[1024];
    const int b = blockIdx.x, tid = threadIdx.x;

    float vals[4];
    float m = -1e30f;
    #pragma unroll
    for (int r = 0; r < 4; ++r) {
        vals[r] = scores[b * Hn + tid + r * 1024];
        m = fmaxf(m, vals[r]);
    }
    red[tid] = m;
    __syncthreads();
    for (int s = 512; s > 0; s >>= 1) {
        if (tid < s) red[tid] = fmaxf(red[tid], red[tid + s]);
        __syncthreads();
    }
    m = red[0];
    __syncthreads();

    float sum = 0.0f;
    #pragma unroll
    for (int r = 0; r < 4; ++r) {
        vals[r] = expf(vals[r] - m);
        sum += vals[r];
    }
    red[tid] = sum;
    __syncthreads();
    for (int s = 512; s > 0; s >>= 1) {
        if (tid < s) red[tid] += red[tid + s];
        __syncthreads();
    }
    float inv = 1.0f / red[0];
    #pragma unroll
    for (int r = 0; r < 4; ++r)
        weights[b * Hn + tid + r * 1024] = vals[r] * inv;
}

// ---------------------------------------------------------------------------
// Phase 4: output[b][d] = sum_h weights[b][h] * cs[b][h][d]   (streams 1 GiB)
// One wave per 16-wide d tile; f32 WMMA B-layout maps directly onto coalesced
// 64B row-segment loads: lanes0-15 -> K0/K1, lanes16-31 -> K2/K3.
// ---------------------------------------------------------------------------
__global__ void __launch_bounds__(256) output_kernel(const float* __restrict__ cs,
                                                     const float* __restrict__ wts,
                                                     float* __restrict__ out) {
    const int tid  = threadIdx.x;
    const int lane = tid & 31;
    const int w    = tid >> 5;
    const int b    = blockIdx.y;
    const int d0   = (blockIdx.x * 8 + w) * 16;
    const int n    = lane & 15;
    const int kb   = (lane < 16) ? 0 : 2;

    const float* base = cs + (size_t)b * Hn * Dn + d0 + n;
    const float* wrow = wts + b * Hn;

#ifdef HAVE_WMMA_F32X4
    v8f acc = {};
    #pragma unroll 4
    for (int h = 0; h < Hn; h += 4) {
        __builtin_prefetch(base + (size_t)(h + 16 + kb) * Dn, 0, 0);
        float b0 = base[(size_t)(h + kb) * Dn];
        float b1 = base[(size_t)(h + kb + 1) * Dn];
        v2f av = {wrow[h + kb], wrow[h + kb + 1]};   // A broadcast across M
        v2f bv = {b0, b1};
        acc = __builtin_amdgcn_wmma_f32_16x16x4_f32(
                  false, av, false, bv, (short)0, acc, false, false);
    }
    if (lane < 16) out[b * Dn + d0 + lane] = acc[0];
#else
    float accs = 0.0f;
    #pragma unroll 4
    for (int h = 0; h < Hn; h += 4) {
        __builtin_prefetch(base + (size_t)(h + 16 + kb) * Dn, 0, 0);
        float b0 = base[(size_t)(h + kb) * Dn];
        float b1 = base[(size_t)(h + kb + 1) * Dn];
        accs += wrow[h + kb] * b0 + wrow[h + kb + 1] * b1;
    }
    accs += __shfl_xor(accs, 16, 32);   // combine K0/K1 half with K2/K3 half
    if (lane < 16) out[b * Dn + d0 + lane] = accs;
#endif
}

// ---------------------------------------------------------------------------
extern "C" void kernel_launch(void* const* d_in, const int* in_sizes, int n_in,
                              void* d_out, int out_size, void* d_ws, size_t ws_size,
                              hipStream_t stream) {
    const float* x  = (const float*)d_in[0];
    const float* cs = (const float*)d_in[1];
    const float* t  = (const float*)d_in[2];
    const float* wq = (const float*)d_in[3];
    const float* bq = (const float*)d_in[4];
    const float* wk = (const float*)d_in[5];
    const float* bk = (const float*)d_in[6];

    float* out     = (float*)d_out;            // [0 .. 65535] = output
    float* weights = out + Bsz * Hn;           // [65536 .. 131071] = weights
    float* ws      = (float*)d_ws;
    float* q2      = ws;                       // 16*256 floats
    float* scores  = ws + Bsz * 256;           // 16*4096 floats

    compute_q_kernel<<<dim3(8), dim3(256), 0, stream>>>(x, t, wq, bq, q2);
    scores_kernel  <<<dim3(32, 16), dim3(256), 0, stream>>>(cs, wk, bk, q2, scores);
    softmax_kernel <<<dim3(16), dim3(1024), 0, stream>>>(scores, weights);
    output_kernel  <<<dim3(32, 16), dim3(256), 0, stream>>>(cs, weights, out);
}
